// LSTMTagger_26328149525252
// MI455X (gfx1250) — matrix operand order repaired and checked by reading, not compiled, not run on hardware
//
#include <hip/hip_runtime.h>
#include <hip/hip_bf16.h>

// ---------------------------------------------------------------------------
// Types for CDNA5 WMMA (gfx1250, wave32)
// ---------------------------------------------------------------------------
typedef __attribute__((ext_vector_type(16))) __bf16 bf16x16;
typedef __attribute__((ext_vector_type(8)))  float  f32x8;

union Frag {
    bf16x16        v;
    unsigned short u[16];
    uint4          q[2];
};

__device__ __forceinline__ f32x8 wmma_bf16(bf16x16 a, bf16x16 b, f32x8 c) {
    // D = A(16x32 bf16) * B(32x16 bf16) + C(16x16 f32)
    return __builtin_amdgcn_wmma_f32_16x16x32_bf16(
        /*neg_a=*/false, a, /*neg_b=*/false, b,
        /*c_mod=*/(short)0, c, /*reuse_a=*/false, /*reuse_b=*/false);
}

__device__ __forceinline__ unsigned short f2bf(float f) {
    unsigned int u = __float_as_uint(f);
    unsigned int r = (u + 0x7FFFu + ((u >> 16) & 1u)) >> 16;  // round-nearest-even
    return (unsigned short)r;
}

__device__ __forceinline__ float sigm(float x) { return 1.f / (1.f + __expf(-x)); }

// ---------------------------------------------------------------------------
// Problem dimensions
// ---------------------------------------------------------------------------
#define S_   4096
#define L_   16
#define WE_  300
#define CE_  30
#define CH_  25
#define H_   512
#define G_   (4 * H_)          // 2048 gates
#define DIN_ (WE_ + 2 * CH_)   // 350
#define KPAD_ 352              // 11 K-tiles of 32
#define KT_X 11
#define NT_X 128               // 2048 / 16
#define KT_H 16                // 512 / 32
#define KT_T 32                // 1024 / 32
#define NT_T 4                 // 64 / 16 (52 padded)
#define NL_  52

// ---------------------------------------------------------------------------
// Pack a row-major weight W (N rows x K cols, gemm computes x @ W^T) into the
// per-lane 16-bit B-fragment layout: dst[((kt*NT+nt)*32+lane)*16+j],
// column n = nt*16 + (lane&15), k = kt*32 + (j<8 ? h*8+j : 16+h*8+(j-8)), h = lane>>4.
// ---------------------------------------------------------------------------
__global__ void k_pack_b(const float* __restrict__ W, unsigned short* __restrict__ dst,
                         int N, int K, int NT, int total) {
    int i = blockIdx.x * blockDim.x + threadIdx.x;
    if (i >= total) return;
    int j    = i & 15;
    int lane = (i >> 4) & 31;
    int tile = i >> 9;
    int nt   = tile % NT;
    int kt   = tile / NT;
    int n = nt * 16 + (lane & 15);
    int k = kt * 32 + ((j < 8) ? ((lane >> 4) * 8 + j) : (16 + (lane >> 4) * 8 + (j - 8)));
    float v = (n < N && k < K) ? W[(size_t)n * K + k] : 0.f;
    dst[i] = f2bf(v);
}

// Combined gate biases for the word LSTM (bih + bhh), both directions.
__global__ void k_bias(const float* bih_f, const float* bhh_f,
                       const float* bih_b, const float* bhh_b,
                       float* bf_, float* bb_) {
    int i = blockIdx.x * blockDim.x + threadIdx.x;
    if (i < G_) { bf_[i] = bih_f[i] + bhh_f[i]; bb_[i] = bih_b[i] + bhh_b[i]; }
}

// Gather word embeddings into the bf16 word_reps buffer (cols 0..299), zero pad.
__global__ void k_reps_word(const int* __restrict__ words,
                            const float* __restrict__ word_emb,
                            unsigned short* __restrict__ reps) {
    long i = (long)blockIdx.x * blockDim.x + threadIdx.x;
    if (i >= (long)S_ * KPAD_) return;
    int w = (int)(i / KPAD_);
    int j = (int)(i % KPAD_);
    float v = 0.f;
    if (j < WE_) v = word_emb[(size_t)words[w] * WE_ + j];
    reps[i] = f2bf(v);     // char columns (300..349) overwritten by k_char later
}

// ---------------------------------------------------------------------------
// Char BiLSTM: one thread per word, weights staged in LDS.
// Writes h_fwd -> reps[.,300..324], h_bwd -> reps[.,325..349] as bf16.
// ---------------------------------------------------------------------------
__device__ __forceinline__ void char_cell(const float* __restrict__ x,
                                          float* h, float* c,
                                          const float* Wih, const float* Whh,
                                          const float* bias) {
    float hn[CH_];
    for (int k = 0; k < CH_; ++k) {
        float gi = bias[k], gf = bias[CH_ + k], gg = bias[2 * CH_ + k], go = bias[3 * CH_ + k];
        for (int j = 0; j < CE_; ++j) {
            float xv = x[j];
            gi += xv * Wih[k * CE_ + j];
            gf += xv * Wih[(CH_ + k) * CE_ + j];
            gg += xv * Wih[(2 * CH_ + k) * CE_ + j];
            go += xv * Wih[(3 * CH_ + k) * CE_ + j];
        }
        for (int j = 0; j < CH_; ++j) {
            float hv = h[j];
            gi += hv * Whh[k * CH_ + j];
            gf += hv * Whh[(CH_ + k) * CH_ + j];
            gg += hv * Whh[(2 * CH_ + k) * CH_ + j];
            go += hv * Whh[(3 * CH_ + k) * CH_ + j];
        }
        c[k]  = sigm(gf) * c[k] + sigm(gi) * tanhf(gg);
        hn[k] = sigm(go) * tanhf(c[k]);
    }
    for (int k = 0; k < CH_; ++k) h[k] = hn[k];
}

__global__ void __launch_bounds__(256)
k_char(const int* __restrict__ chars, const int* __restrict__ clen,
       const float* __restrict__ char_emb,
       const float* Wihf, const float* Whhf, const float* bihf, const float* bhhf,
       const float* Wihb, const float* Whhb, const float* bihb, const float* bhhb,
       unsigned short* __restrict__ reps) {
    __shared__ float sWihF[4 * CH_ * CE_], sWhhF[4 * CH_ * CH_], sBF[4 * CH_];
    __shared__ float sWihB[4 * CH_ * CE_], sWhhB[4 * CH_ * CH_], sBB[4 * CH_];
    int tid = threadIdx.x;
    for (int i = tid; i < 4 * CH_ * CE_; i += 256) { sWihF[i] = Wihf[i]; sWihB[i] = Wihb[i]; }
    for (int i = tid; i < 4 * CH_ * CH_; i += 256) { sWhhF[i] = Whhf[i]; sWhhB[i] = Whhb[i]; }
    for (int i = tid; i < 4 * CH_; i += 256)       { sBF[i] = bihf[i] + bhhf[i]; sBB[i] = bihb[i] + bhhb[i]; }
    __syncthreads();

    int w = blockIdx.x * 256 + tid;
    if (w >= S_) return;
    int len = clen[w];
    int idx[L_];
    for (int t = 0; t < L_; ++t) idx[t] = chars[w * L_ + t];

    float h[CH_], c[CH_];
    // forward (masked: only steps t < len update state)
    for (int k = 0; k < CH_; ++k) { h[k] = 0.f; c[k] = 0.f; }
    for (int t = 0; t < L_; ++t)
        if (t < len) char_cell(char_emb + (size_t)idx[t] * CE_, h, c, sWihF, sWhhF, sBF);
    for (int k = 0; k < CH_; ++k) reps[(size_t)w * KPAD_ + WE_ + k] = f2bf(h[k]);
    // backward (reversed within valid length)
    for (int k = 0; k < CH_; ++k) { h[k] = 0.f; c[k] = 0.f; }
    for (int t = 0; t < L_; ++t)
        if (t < len) char_cell(char_emb + (size_t)idx[len - 1 - t] * CE_, h, c, sWihB, sWhhB, sBB);
    for (int k = 0; k < CH_; ++k) reps[(size_t)w * KPAD_ + WE_ + CH_ + k] = f2bf(h[k]);
}

// ---------------------------------------------------------------------------
// X-part GEMM: X[dir][t][g] = word_reps[t] @ Wih[dir]^T + (bih+bhh)[dir]
// One wave per 16x16 output tile, bf16 WMMA over 11 K-tiles.
// ---------------------------------------------------------------------------
__global__ void __launch_bounds__(256)
k_xgemm(const unsigned short* __restrict__ reps,
        const unsigned short* __restrict__ wihf, const unsigned short* __restrict__ wihb,
        const float* __restrict__ biasf, const float* __restrict__ biasb,
        float* __restrict__ Xf, float* __restrict__ Xb) {
    int wave = (blockIdx.x * blockDim.x + threadIdx.x) >> 5;
    int lane = threadIdx.x & 31;
    int dir  = wave >> 15;                 // 32768 tiles per direction
    int r    = wave & 32767;
    int mt   = r >> 7;                     // 256 M-tiles
    int nt   = r & 127;                    // 128 N-tiles
    const unsigned short* wp = dir ? wihb : wihf;
    const float* bp = dir ? biasb : biasf;
    float* X = dir ? Xb : Xf;
    int half = lane >> 4;

    f32x8 acc = {};
    for (int kt = 0; kt < KT_X; ++kt) {
        Frag a, b;
        // A fragment: row m, K = kt*32 + {half*8..+7, 16+half*8..+7}
        const uint4* pa = reinterpret_cast<const uint4*>(
            reps + (size_t)(mt * 16 + (lane & 15)) * KPAD_ + kt * 32 + half * 8);
        a.q[0] = pa[0];
        a.q[1] = pa[2];
        const uint4* pb = reinterpret_cast<const uint4*>(
            wp + ((size_t)(kt * NT_X + nt) * 32 + lane) * 16);
        b.q[0] = pb[0]; b.q[1] = pb[1];
        acc = wmma_bf16(a.v, b.v, acc);
    }
    int n  = nt * 16 + (lane & 15);
    int mb = mt * 16 + half * 8;
    float bias = bp[n];
    for (int j = 0; j < 8; ++j)
        X[(size_t)(mb + j) * G_ + n] = acc[j] + bias;
}

// ---------------------------------------------------------------------------
// Sequential recurrence: one persistent 1024-thread block per direction.
// Per step:
//   - WMMA matvec h @ Whh^T (h broadcast into all 16 A rows, row 0 of D used)
//   - the NEXT step's precomputed X row is prefetched into an LDS double
//     buffer with global_load_async_to_lds_b64 (ASYNCcnt path), overlapping
//     the matvec; retired with s_wait_asynccnt + workgroup barrier
//   - elementwise cell update (c lives in registers of threads 0..511)
// ---------------------------------------------------------------------------
__global__ void __launch_bounds__(1024)
k_word_recurrent(const float* __restrict__ Xf, const float* __restrict__ Xb,
                 const unsigned short* __restrict__ whhf,
                 const unsigned short* __restrict__ whhb,
                 float* __restrict__ hsf, float* __restrict__ hsb) {
    __shared__ __align__(16) unsigned short hbf[H_];
    __shared__ float gates[G_];
    __shared__ __align__(16) float xstage[2][G_];   // 16 KB double buffer
    const int dir = blockIdx.x;
    const float* Xpre = dir ? Xb : Xf;
    const unsigned short* wp = dir ? whhb : whhf;
    float* hs = dir ? hsb : hsf;
    const int tid = threadIdx.x;
    const int lane = tid & 31, wv = tid >> 5, half = lane >> 4;

    float creg = 0.f;
    if (tid < H_) hbf[tid] = 0;
    __syncthreads();

    // Prologue: async-stage X row for step 0 into buffer 0.
    {
        const int t0 = dir ? (S_ - 1) : 0;
        unsigned lds = (unsigned)(uintptr_t)(&xstage[0][0]) + (unsigned)tid * 8u;
        unsigned long long ga = (unsigned long long)(uintptr_t)(Xpre + (size_t)t0 * G_)
                              + (unsigned long long)tid * 8ull;
        asm volatile("global_load_async_to_lds_b64 %0, %1, off"
                     :: "v"(lds), "v"(ga) : "memory");
    }

    for (int s = 0; s < S_; ++s) {
        const int t   = dir ? (S_ - 1 - s) : s;
        const int buf = s & 1;

        // ---- matvec: gates_pre = h @ Whh^T (overlaps in-flight async copy) --
        f32x8 acc[4] = {};
        #pragma unroll 4
        for (int kt = 0; kt < KT_H; ++kt) {
            Frag a;
            const uint4* pa = reinterpret_cast<const uint4*>(&hbf[kt * 32 + half * 8]);
            a.q[0] = pa[0];
            a.q[1] = pa[2];
            #pragma unroll
            for (int q = 0; q < 4; ++q) {
                const int nt = wv * 4 + q;           // 32 waves x 4 = 128 N-tiles
                Frag b;
                const uint4* pb = reinterpret_cast<const uint4*>(
                    wp + ((size_t)(kt * NT_X + nt) * 32 + lane) * 16);
                b.q[0] = pb[0]; b.q[1] = pb[1];
                acc[q] = wmma_bf16(a.v, b.v, acc[q]);
            }
        }

        // ---- retire async copy of X[t], then combine in LDS ----------------
        asm volatile("s_wait_asynccnt 0x0" ::: "memory");
        __syncthreads();
        if (lane < 16) {
            #pragma unroll
            for (int q = 0; q < 4; ++q) {
                const int n = (wv * 4 + q) * 16 + lane;
                gates[n] = acc[q][0] + xstage[buf][n];   // row M=0 of D
            }
        }

        // ---- issue async stage of the NEXT step's X row (other buffer) -----
        if (s + 1 < S_) {
            const int tn = dir ? (S_ - 2 - s) : (s + 1);
            unsigned lds = (unsigned)(uintptr_t)(&xstage[buf ^ 1][0]) + (unsigned)tid * 8u;
            unsigned long long ga = (unsigned long long)(uintptr_t)(Xpre + (size_t)tn * G_)
                                  + (unsigned long long)tid * 8ull;
            asm volatile("global_load_async_to_lds_b64 %0, %1, off"
                         :: "v"(lds), "v"(ga) : "memory");
        }

        __syncthreads();
        // ---- elementwise LSTM cell update ----------------------------------
        if (tid < H_) {
            float gi = gates[tid], gf = gates[H_ + tid];
            float gg = gates[2 * H_ + tid], go = gates[3 * H_ + tid];
            creg = sigm(gf) * creg + sigm(gi) * tanhf(gg);
            float hv = sigm(go) * tanhf(creg);
            hs[(size_t)t * H_ + tid] = hv;
            hbf[tid] = f2bf(hv);
        }
        __syncthreads();
    }
}

// ---------------------------------------------------------------------------
// Tag projection: out = concat(hs_f, hs_b) @ tag_W^T + tag_b  (4096 x 52)
// ---------------------------------------------------------------------------
__global__ void __launch_bounds__(256)
k_tag(const float* __restrict__ hsf, const float* __restrict__ hsb,
      const unsigned short* __restrict__ tagw, const float* __restrict__ tag_b,
      float* __restrict__ out) {
    int wave = (blockIdx.x * blockDim.x + threadIdx.x) >> 5;
    int lane = threadIdx.x & 31;
    int mt = wave >> 2;                    // 256 M-tiles
    int nt = wave & 3;                     // 4 N-tiles (52 padded to 64)
    int half = lane >> 4;
    int m = mt * 16 + (lane & 15);

    f32x8 acc = {};
    for (int kt = 0; kt < KT_T; ++kt) {
        Frag a, b;
        #pragma unroll
        for (int j = 0; j < 16; ++j) {
            int k = kt * 32 + ((j < 8) ? (half * 8 + j) : (16 + half * 8 + (j - 8)));
            float v = (k < H_) ? hsf[(size_t)m * H_ + k] : hsb[(size_t)m * H_ + (k - H_)];
            a.u[j] = f2bf(v);
        }
        const uint4* pb = reinterpret_cast<const uint4*>(
            tagw + ((size_t)(kt * NT_T + nt) * 32 + lane) * 16);
        b.q[0] = pb[0]; b.q[1] = pb[1];
        acc = wmma_bf16(a.v, b.v, acc);
    }
    int n  = nt * 16 + (lane & 15);
    int mb = mt * 16 + half * 8;
    if (n < NL_) {
        float tb = tag_b[n];
        for (int j = 0; j < 8; ++j)
            out[(size_t)(mb + j) * NL_ + n] = acc[j] + tb;
    }
}

// ---------------------------------------------------------------------------
// Host-side orchestration
// ---------------------------------------------------------------------------
extern "C" void kernel_launch(void* const* d_in, const int* in_sizes, int n_in,
                              void* d_out, int out_size, void* d_ws, size_t ws_size,
                              hipStream_t stream) {
    (void)in_sizes; (void)n_in; (void)out_size; (void)ws_size;

    const int*   words    = (const int*)  d_in[0];
    const int*   chars    = (const int*)  d_in[1];
    const int*   clen     = (const int*)  d_in[2];
    const float* word_emb = (const float*)d_in[3];
    const float* char_emb = (const float*)d_in[4];
    const float* cWih_f = (const float*)d_in[5],  *cWhh_f = (const float*)d_in[6];
    const float* cbih_f = (const float*)d_in[7],  *cbhh_f = (const float*)d_in[8];
    const float* cWih_b = (const float*)d_in[9],  *cWhh_b = (const float*)d_in[10];
    const float* cbih_b = (const float*)d_in[11], *cbhh_b = (const float*)d_in[12];
    const float* wWih_f = (const float*)d_in[13], *wWhh_f = (const float*)d_in[14];
    const float* wbih_f = (const float*)d_in[15], *wbhh_f = (const float*)d_in[16];
    const float* wWih_b = (const float*)d_in[17], *wWhh_b = (const float*)d_in[18];
    const float* wbih_b = (const float*)d_in[19], *wbhh_b = (const float*)d_in[20];
    const float* tag_W  = (const float*)d_in[21], *tag_b  = (const float*)d_in[22];
    float* out = (float*)d_out;

    // Workspace layout (256B aligned slabs)
    char* base = (char*)d_ws;
    size_t off = 0;
    auto alloc = [&](size_t bytes) -> char* {
        char* p = base + off;
        off = (off + bytes + 255) & ~(size_t)255;
        return p;
    };
    const size_t WIH_PACK = (size_t)KT_X * NT_X * 32 * 16 * 2;  // 1.44 MB
    const size_t WHH_PACK = (size_t)KT_H * NT_X * 32 * 16 * 2;  // 2 MB
    const size_t TAG_PACK = (size_t)KT_T * NT_T * 32 * 16 * 2;  // 128 KB
    unsigned short* p_wihf = (unsigned short*)alloc(WIH_PACK);
    unsigned short* p_wihb = (unsigned short*)alloc(WIH_PACK);
    unsigned short* p_whhf = (unsigned short*)alloc(WHH_PACK);
    unsigned short* p_whhb = (unsigned short*)alloc(WHH_PACK);
    unsigned short* p_tagw = (unsigned short*)alloc(TAG_PACK);
    float* p_biasf = (float*)alloc(G_ * sizeof(float));
    float* p_biasb = (float*)alloc(G_ * sizeof(float));
    unsigned short* p_reps = (unsigned short*)alloc((size_t)S_ * KPAD_ * 2);
    float* p_Xf  = (float*)alloc((size_t)S_ * G_ * sizeof(float));
    float* p_Xb  = (float*)alloc((size_t)S_ * G_ * sizeof(float));
    float* p_hsf = (float*)alloc((size_t)S_ * H_ * sizeof(float));
    float* p_hsb = (float*)alloc((size_t)S_ * H_ * sizeof(float));

    // 1) Pack weights into WMMA B-fragment layouts (bf16)
    {
        int tot = KT_X * NT_X * 512;
        k_pack_b<<<(tot + 255) / 256, 256, 0, stream>>>(wWih_f, p_wihf, G_, DIN_, NT_X, tot);
        k_pack_b<<<(tot + 255) / 256, 256, 0, stream>>>(wWih_b, p_wihb, G_, DIN_, NT_X, tot);
    }
    {
        int tot = KT_H * NT_X * 512;
        k_pack_b<<<(tot + 255) / 256, 256, 0, stream>>>(wWhh_f, p_whhf, G_, H_, NT_X, tot);
        k_pack_b<<<(tot + 255) / 256, 256, 0, stream>>>(wWhh_b, p_whhb, G_, H_, NT_X, tot);
    }
    {
        int tot = KT_T * NT_T * 512;
        k_pack_b<<<(tot + 255) / 256, 256, 0, stream>>>(tag_W, p_tagw, NL_, 2 * H_, NT_T, tot);
    }
    k_bias<<<(G_ + 255) / 256, 256, 0, stream>>>(wbih_f, wbhh_f, wbih_b, wbhh_b, p_biasf, p_biasb);

    // 2) word_reps: embedding gather (bf16) then char BiLSTM fills cols 300..349
    {
        long tot = (long)S_ * KPAD_;
        k_reps_word<<<(int)((tot + 255) / 256), 256, 0, stream>>>(words, word_emb, p_reps);
    }
    k_char<<<S_ / 256, 256, 0, stream>>>(chars, clen, char_emb,
                                         cWih_f, cWhh_f, cbih_f, cbhh_f,
                                         cWih_b, cWhh_b, cbih_b, cbhh_b, p_reps);

    // 3) Hoisted input GEMM for both directions (WMMA bf16)
    k_xgemm<<<(2 * 256 * 128) / 8, 256, 0, stream>>>(p_reps, p_wihf, p_wihb,
                                                     p_biasf, p_biasb, p_Xf, p_Xb);

    // 4) Sequential bidirectional recurrence (persistent block per direction,
    //    async-LDS double buffering of the precomputed X rows)
    k_word_recurrent<<<2, 1024, 0, stream>>>(p_Xf, p_Xb, p_whhf, p_whhb, p_hsf, p_hsb);

    // 5) Tag projection (WMMA bf16)
    k_tag<<<(256 * NT_T) / 8, 256, 0, stream>>>(p_hsf, p_hsb, p_tagw, tag_b, out);
}
